// MultiheadSelfAttention_62620623175755
// MI455X (gfx1250) — matrix operand order repaired and checked by reading, not compiled, not run on hardware
//
#include <hip/hip_runtime.h>

#define B_  2
#define S_  4096
#define D_  512
#define H_  8
#define HD_ 64
#define M_  (B_ * S_)   // 8192
#define TK  32          // keys per tile

typedef __attribute__((ext_vector_type(16))) __bf16        v16bf;
typedef __attribute__((ext_vector_type(8)))  float         v8f;
typedef __attribute__((ext_vector_type(4)))  unsigned int  v4u;

union FragBF { v16bf v; v4u q[2]; };

__device__ __forceinline__ unsigned short f2bf(float f) {
  unsigned int u = __float_as_uint(f);
  u += 0x7FFFu + ((u >> 16) & 1u);          // round-to-nearest-even
  return (unsigned short)(u >> 16);
}

// A fragment: 16x32 bf16 tile at (row0, k0) of row-major src (leading dim ld).
__device__ __forceinline__ v16bf load_frag_A(const unsigned short* __restrict__ src,
                                             int ld, int row0, int k0, int lane) {
  const int half = lane >> 4, row = lane & 15;
  const unsigned short* p = src + (size_t)(row0 + row) * ld + k0 + half * 8;
  FragBF f;
  f.q[0] = *(const v4u*)(p);        // K rel 0..7
  f.q[1] = *(const v4u*)(p + 16);   // K rel 16..23
  return f.v;
}

// B fragment: 32x16 (KxN) where column n of B is row (n0+n) of row-major src.
// Works for global or LDS pointers (addrspace inferred -> ds_load_b128 for LDS).
__device__ __forceinline__ v16bf load_frag_B(const unsigned short* src,
                                             int ld, int n0, int k0, int lane) {
  const int half = lane >> 4, col = lane & 15;
  const unsigned short* p = src + (size_t)(n0 + col) * ld + k0 + half * 8;
  FragBF f;
  f.q[0] = *(const v4u*)(p);        // K rel 0..7
  f.q[1] = *(const v4u*)(p + 16);   // K rel 16..23
  return f.v;
}

// CDNA5 async copy: 16B per lane, global -> LDS, tracked with ASYNCcnt.
__device__ __forceinline__ void async_b128(unsigned ldsoff, const void* gaddr) {
  asm volatile("global_load_async_to_lds_b128 %0, %1, off"
               :: "v"(ldsoff), "v"(gaddr) : "memory");
}
__device__ __forceinline__ void wait_async_le4() {
  asm volatile("s_wait_asynccnt 0x4" ::: "memory");
}
__device__ __forceinline__ unsigned lds_off(const void* p) {
  return (unsigned)(uintptr_t)p;    // LDS aperture: low 32 bits = LDS byte offset
}

__global__ void cvt_f32_bf16(const float* __restrict__ src,
                             unsigned short* __restrict__ dst, int n) {
  int i = blockIdx.x * blockDim.x + threadIdx.x;
  if (i < n) dst[i] = f2bf(src[i]);
}

// Unified GEMM: Y[m,n] = sum_k A[m,k] * W[n,k] + bias[n]
// mode 0/1: bf16 -> [B,H,S,HD]   mode 2: bf16 -> V^T [B,H,HD,S]   mode 3: f32 -> [M,D]
__global__ __launch_bounds__(128)
void gemm_nt(const unsigned short* __restrict__ A,
             const unsigned short* __restrict__ W,
             const float* __restrict__ bias,
             unsigned short* __restrict__ obf,
             float* __restrict__ of32, int mode) {
  const int lane = threadIdx.x & 31;
  const int wv   = threadIdx.x >> 5;
  const int m0   = blockIdx.x * 64 + (wv >> 1) * 32;
  const int n0   = blockIdx.y * 64 + (wv & 1) * 32;

  v8f c[2][2] = {};
#pragma unroll 4
  for (int k = 0; k < D_; k += 32) {
    v16bf a0 = load_frag_A(A, D_, m0,      k, lane);
    v16bf a1 = load_frag_A(A, D_, m0 + 16, k, lane);
    v16bf b0 = load_frag_B(W, D_, n0,      k, lane);
    v16bf b1 = load_frag_B(W, D_, n0 + 16, k, lane);
    c[0][0] = __builtin_amdgcn_wmma_f32_16x16x32_bf16(false, a0, false, b0, (short)0, c[0][0], false, false);
    c[0][1] = __builtin_amdgcn_wmma_f32_16x16x32_bf16(false, a0, false, b1, (short)0, c[0][1], false, false);
    c[1][0] = __builtin_amdgcn_wmma_f32_16x16x32_bf16(false, a1, false, b0, (short)0, c[1][0], false, false);
    c[1][1] = __builtin_amdgcn_wmma_f32_16x16x32_bf16(false, a1, false, b1, (short)0, c[1][1], false, false);
  }

  const int half = lane >> 4, nl = lane & 15;
#pragma unroll
  for (int i = 0; i < 2; ++i)
#pragma unroll
    for (int j = 0; j < 2; ++j) {
      const int n  = n0 + j * 16 + nl;
      const float bn = bias[n];
      const int h = n >> 6, hd = n & 63;
#pragma unroll
      for (int r = 0; r < 8; ++r) {
        const int m = m0 + i * 16 + r + half * 8;
        const float val = c[i][j][r] + bn;
        if (mode == 3) {
          of32[(size_t)m * D_ + n] = val;
        } else {
          const int b = m >> 12, s = m & (S_ - 1);
          if (mode == 2)
            obf[((size_t)((b * H_ + h) * HD_ + hd)) * S_ + s] = f2bf(val);  // V^T
          else
            obf[((size_t)(b * H_ + h) * S_ + s) * HD_ + hd] = f2bf(val);    // Q/K
        }
      }
    }
}

// Flash attention: grid (B*H, S/64), 128 threads (4 waves), wave owns 16 query rows.
// K/V tiles staged once per WG into LDS via async loads, double-buffered.
__global__ __launch_bounds__(128)
void flash_attn(const unsigned short* __restrict__ qb,
                const unsigned short* __restrict__ kb,
                const unsigned short* __restrict__ vt,
                unsigned short* __restrict__ attn) {
  __shared__ unsigned short Klds[2][TK * HD_];   // [32 keys][64 hd]
  __shared__ unsigned short Vlds[2][HD_ * TK];   // [64 hd][32 keys]
  __shared__ unsigned short Plds[4][16 * TK];    // per-wave P tile

  const int tid  = threadIdx.x;
  const int lane = tid & 31;
  const int wv   = tid >> 5;
  const int bh   = blockIdx.x;                    // b*H + h
  const int m0   = blockIdx.y * 64 + wv * 16;

  const unsigned short* q = qb + (size_t)bh * S_ * HD_;
  const unsigned short* k = kb + (size_t)bh * S_ * HD_;
  const unsigned short* v = vt + (size_t)bh * HD_ * S_;

  // Per-thread async-copy slots: 256 x 16B chunks per 4KB tile, 2 per thread.
  const unsigned kbase0 = lds_off(&Klds[0][0]), kbase1 = lds_off(&Klds[1][0]);
  const unsigned vbase0 = lds_off(&Vlds[0][0]), vbase1 = lds_off(&Vlds[1][0]);
  const int c0 = tid, c1 = tid + 128;
  const int vrow0 = c0 >> 2, vcol0 = (c0 & 3) * 16;   // V: row=hd (64B rows)
  const int vrow1 = c1 >> 2, vcol1 = (c1 & 3) * 16;

  auto prefetch = [&](int t, int buf) {
    const unsigned kb_ = buf ? kbase1 : kbase0;
    const unsigned vb_ = buf ? vbase1 : vbase0;
    // K tile: rows t*32..t*32+31 are contiguous 4KB (ld == 64)
    const char* kg = (const char*)(k + (size_t)t * TK * HD_);
    async_b128(kb_ + c0 * 16, kg + c0 * 16);
    async_b128(kb_ + c1 * 16, kg + c1 * 16);
    // V^T tile: 64 rows of 64B, global row stride S_*2 bytes
    const char* vg = (const char*)(v + (size_t)t * TK);
    async_b128(vb_ + c0 * 16, vg + (size_t)vrow0 * (S_ * 2) + vcol0);
    async_b128(vb_ + c1 * 16, vg + (size_t)vrow1 * (S_ * 2) + vcol1);
  };

  const v16bf qf0 = load_frag_A(q, HD_, m0, 0,  lane);
  const v16bf qf1 = load_frag_A(q, HD_, m0, 32, lane);

  v8f o[4] = {};
  float mrow[8], lrow[8];
#pragma unroll
  for (int r = 0; r < 8; ++r) { mrow[r] = -1e30f; lrow[r] = 0.f; }

  const float scale = 0.125f;                     // 1/sqrt(64)
  const int half = lane >> 4, nl = lane & 15;
  unsigned short* pl = Plds[wv];

  const int NT = S_ / TK;                         // 128
  prefetch(0, 0);

  for (int t = 0; t < NT; ++t) {
    const int nt = (t + 1 == NT) ? 0 : t + 1;     // wrap: harmless extra prefetch
    prefetch(nt, (t + 1) & 1);
    wait_async_le4();                             // own wave's tile-t loads done
    __syncthreads();                              // all waves' portions done

    const unsigned short* Kt = Klds[t & 1];
    const unsigned short* Vt = Vlds[t & 1];

    // ---- scores: S[16 x 32] = Q[16x64] @ K^T ----
    v8f s0 = {}, s1 = {};
    {
      v16bf k0f = load_frag_B(Kt, HD_, 0,  0,  lane);
      v16bf k1f = load_frag_B(Kt, HD_, 0,  32, lane);
      s0 = __builtin_amdgcn_wmma_f32_16x16x32_bf16(false, qf0, false, k0f, (short)0, s0, false, false);
      s0 = __builtin_amdgcn_wmma_f32_16x16x32_bf16(false, qf1, false, k1f, (short)0, s0, false, false);
      v16bf k2f = load_frag_B(Kt, HD_, 16, 0,  lane);
      v16bf k3f = load_frag_B(Kt, HD_, 16, 32, lane);
      s1 = __builtin_amdgcn_wmma_f32_16x16x32_bf16(false, qf0, false, k2f, (short)0, s1, false, false);
      s1 = __builtin_amdgcn_wmma_f32_16x16x32_bf16(false, qf1, false, k3f, (short)0, s1, false, false);
    }

    // ---- online softmax (row = r + 8*half across 16-lane groups) ----
    float pmax[8];
#pragma unroll
    for (int r = 0; r < 8; ++r) {
      s0[r] *= scale; s1[r] *= scale;
      float tmx = fmaxf(s0[r], s1[r]);
      tmx = fmaxf(tmx, __shfl_xor(tmx, 1, 16));
      tmx = fmaxf(tmx, __shfl_xor(tmx, 2, 16));
      tmx = fmaxf(tmx, __shfl_xor(tmx, 4, 16));
      tmx = fmaxf(tmx, __shfl_xor(tmx, 8, 16));
      pmax[r] = tmx;
    }
#pragma unroll
    for (int r = 0; r < 8; ++r) {
      const float mnew  = fmaxf(mrow[r], pmax[r]);
      const float alpha = __expf(mrow[r] - mnew);
      const float p0 = __expf(s0[r] - mnew);
      const float p1 = __expf(s1[r] - mnew);
      s0[r] = p0; s1[r] = p1;
      float ps = p0 + p1;
      ps += __shfl_xor(ps, 1, 16);
      ps += __shfl_xor(ps, 2, 16);
      ps += __shfl_xor(ps, 4, 16);
      ps += __shfl_xor(ps, 8, 16);
      lrow[r] = lrow[r] * alpha + ps;
      mrow[r] = mnew;
#pragma unroll
      for (int j = 0; j < 4; ++j) o[j][r] *= alpha;
    }

    // ---- P: C layout -> LDS -> bf16 A fragment (wave-private, dscnt-ordered) ----
#pragma unroll
    for (int r = 0; r < 8; ++r) {
      const int row = r + half * 8;
      pl[row * TK + nl]      = f2bf(s0[r]);
      pl[row * TK + 16 + nl] = f2bf(s1[r]);
    }
    FragBF pf;
    const unsigned short* pp = pl + nl * TK + half * 8;
    pf.q[0] = *(const v4u*)(pp);
    pf.q[1] = *(const v4u*)(pp + 16);

    // ---- O += P[16x32] @ V[32x64] ----
#pragma unroll
    for (int j = 0; j < 4; ++j) {
      v16bf vf = load_frag_B(Vt, TK, j * 16, 0, lane);
      o[j] = __builtin_amdgcn_wmma_f32_16x16x32_bf16(false, pf.v, false, vf, (short)0, o[j], false, false);
    }
    __syncthreads();   // all waves done with buf[t&1] before it is refilled
  }

  // ---- normalize, store attn [B,S,D] bf16 ----
  const int b = bh >> 3, h = bh & 7;
#pragma unroll
  for (int j = 0; j < 4; ++j) {
    const int n = j * 16 + nl;
#pragma unroll
    for (int r = 0; r < 8; ++r) {
      const int s = m0 + r + half * 8;
      attn[((size_t)(b * S_ + s)) * D_ + h * HD_ + n] = f2bf(o[j][r] / lrow[r]);
    }
  }
}

extern "C" void kernel_launch(void* const* d_in, const int* in_sizes, int n_in,
                              void* d_out, int out_size, void* d_ws, size_t ws_size,
                              hipStream_t stream) {
  const float* x  = (const float*)d_in[0];
  const float* Wq = (const float*)d_in[1];
  const float* bq = (const float*)d_in[2];
  const float* Wk = (const float*)d_in[3];
  const float* bk = (const float*)d_in[4];
  const float* Wv = (const float*)d_in[5];
  const float* bv = (const float*)d_in[6];
  const float* Wo = (const float*)d_in[7];
  const float* bo = (const float*)d_in[8];

  char* ws = (char*)d_ws;
  size_t off = 0;
  auto alloc = [&](size_t bytes) {
    char* p = ws + off;
    off += (bytes + 255) & ~(size_t)255;
    return p;
  };
  unsigned short* x_bf  = (unsigned short*)alloc((size_t)M_ * D_ * 2);  // reused as attn_bf
  unsigned short* wq_bf = (unsigned short*)alloc((size_t)D_ * D_ * 2);
  unsigned short* wk_bf = (unsigned short*)alloc((size_t)D_ * D_ * 2);
  unsigned short* wv_bf = (unsigned short*)alloc((size_t)D_ * D_ * 2);
  unsigned short* wo_bf = (unsigned short*)alloc((size_t)D_ * D_ * 2);
  unsigned short* q_buf = (unsigned short*)alloc((size_t)M_ * D_ * 2);
  unsigned short* k_buf = (unsigned short*)alloc((size_t)M_ * D_ * 2);
  unsigned short* v_t   = (unsigned short*)alloc((size_t)M_ * D_ * 2);
  unsigned short* attn_bf = x_bf;   // x_bf dead after QKV GEMMs

  const int nx = M_ * D_;
  const int nw = D_ * D_;
  cvt_f32_bf16<<<(nx + 255) / 256, 256, 0, stream>>>(x,  x_bf,  nx);
  cvt_f32_bf16<<<(nw + 255) / 256, 256, 0, stream>>>(Wq, wq_bf, nw);
  cvt_f32_bf16<<<(nw + 255) / 256, 256, 0, stream>>>(Wk, wk_bf, nw);
  cvt_f32_bf16<<<(nw + 255) / 256, 256, 0, stream>>>(Wv, wv_bf, nw);
  cvt_f32_bf16<<<(nw + 255) / 256, 256, 0, stream>>>(Wo, wo_bf, nw);

  dim3 gg(M_ / 64, D_ / 64);  // 128 x 8 workgroups, 4 waves each
  gemm_nt<<<gg, 128, 0, stream>>>(x_bf, wq_bf, bq, q_buf, nullptr, 0);
  gemm_nt<<<gg, 128, 0, stream>>>(x_bf, wk_bf, bk, k_buf, nullptr, 1);
  gemm_nt<<<gg, 128, 0, stream>>>(x_bf, wv_bf, bv, v_t,   nullptr, 2);

  flash_attn<<<dim3(B_ * H_, S_ / 64), 128, 0, stream>>>(q_buf, k_buf, v_t, attn_bf);

  gemm_nt<<<gg, 128, 0, stream>>>(attn_bf, wo_bf, bo, nullptr, (float*)d_out, 3);
}
